// GraphEncoder_75196287418938
// MI455X (gfx1250) — matrix operand order repaired and checked by reading, not compiled, not run on hardware
//
#include <hip/hip_runtime.h>
#include <hip/hip_bf16.h>
#include <math.h>

#define DD 256          // node feature width (H*C)
#define HH 4
#define NEG_SLOPE 0.2f
#define LN_EPS 1e-5f

typedef __attribute__((ext_vector_type(16))) __bf16 v16bf;
typedef __attribute__((ext_vector_type(8)))  float  v8f;

__device__ __forceinline__ __bf16 f2bf(float f) { return (__bf16)f; }

// ---------------------------------------------------------------------------
// Pack W [K x 256] fp32 (row-major, K-major) into bf16 WMMA B-fragments:
// layout [ntile(16)][kstep(K/32)][lane(32)][elem(16)], elem e of lane l is
// B(K = ks*32 + (l<16?0:16) + e, N = nt*16 + (l&15)).
// ---------------------------------------------------------------------------
__global__ void pack_w_kernel(const float* __restrict__ W,
                              __bf16* __restrict__ Wp, int K) {
  const int ksteps = K >> 5;
  const int idx = blockIdx.x * blockDim.x + threadIdx.x;  // (nt,ks,lane)
  const int total = 16 * ksteps * 32;
  if (idx >= total) return;
  const int lane = idx & 31;
  const int ks   = (idx >> 5) % ksteps;
  const int nt   = (idx >> 5) / ksteps;
  const int n     = nt * 16 + (lane & 15);
  const int kbase = ks * 32 + ((lane < 16) ? 0 : 16);
  __bf16* out = Wp + (size_t)idx * 16;
#pragma unroll
  for (int e = 0; e < 16; ++e)
    out[e] = f2bf(W[(size_t)(kbase + e) * DD + n]);
}

// ---------------------------------------------------------------------------
// Y[n_rows x 256] = act( X[n_rows x K] @ W + bias ), K = 32*KSTEPS.
// One wave per 16-row strip; wave cvt-loads its 16xK A-tile to bf16 frags
// once, then streams packed B fragments (L2-resident). N-tile loop unrolled
// x2 -> two independent WMMA chains. Uniform full-tile store fast path.
// ---------------------------------------------------------------------------
template <int KSTEPS>
__global__ void gemm_wmma_kernel(const float* __restrict__ X, int ldx,
                                 const __bf16* __restrict__ Wp,
                                 const float* __restrict__ bias,
                                 float* __restrict__ Y,
                                 int n_rows, int do_relu) {
  const int lane = threadIdx.x & 31;
  const int wave = threadIdx.x >> 5;
  const int rowbase = (blockIdx.x * 8 + wave) * 16;
  if (rowbase >= n_rows) return;

  int arow = rowbase + (lane & 15);
  if (arow >= n_rows) arow = n_rows - 1;         // clamp: tail rows masked at store
  const int koff = (lane < 16) ? 0 : 16;         // A-matrix: contiguous k = 16h + e (matches packed B)
  const float* __restrict__ xrow = X + (size_t)arow * ldx;

  v16bf afrag[KSTEPS];
#pragma unroll
  for (int ks = 0; ks < KSTEPS; ++ks) {
    const int kb = ks * 32 + koff;
#pragma unroll
    for (int i = 0; i < 8; ++i) afrag[ks][i]     = f2bf(xrow[kb + i]);
#pragma unroll
    for (int i = 0; i < 8; ++i) afrag[ks][8 + i] = f2bf(xrow[kb + 8 + i]);
  }

  const v16bf* __restrict__ wp = (const v16bf*)Wp;
  const int mhi  = (lane < 16) ? 0 : 8;          // C/D layout: VGPR r -> M = mhi + r
  const bool full = (rowbase + 16 <= n_rows);    // wave-uniform store fast path

  for (int nt = 0; nt < 16; nt += 2) {
    v8f acc0 = {0.f, 0.f, 0.f, 0.f, 0.f, 0.f, 0.f, 0.f};
    v8f acc1 = {0.f, 0.f, 0.f, 0.f, 0.f, 0.f, 0.f, 0.f};
#pragma unroll
    for (int ks = 0; ks < KSTEPS; ++ks) {
      v16bf b0 = wp[((nt + 0) * KSTEPS + ks) * 32 + lane];
      v16bf b1 = wp[((nt + 1) * KSTEPS + ks) * 32 + lane];
      acc0 = __builtin_amdgcn_wmma_f32_16x16x32_bf16(
          false, afrag[ks], false, b0, (short)0, acc0, false, false);
      acc1 = __builtin_amdgcn_wmma_f32_16x16x32_bf16(
          false, afrag[ks], false, b1, (short)0, acc1, false, false);
    }
    const int col0 = (nt + 0) * 16 + (lane & 15);
    const int col1 = (nt + 1) * 16 + (lane & 15);
    const float bc0 = bias[col0];
    const float bc1 = bias[col1];
    if (full) {
#pragma unroll
      for (int r = 0; r < 8; ++r) {
        const size_t row = (size_t)(rowbase + mhi + r);
        float v0 = acc0[r] + bc0;
        float v1 = acc1[r] + bc1;
        if (do_relu) { v0 = fmaxf(v0, 0.f); v1 = fmaxf(v1, 0.f); }
        Y[row * DD + col0] = v0;
        Y[row * DD + col1] = v1;
      }
    } else {
#pragma unroll
      for (int r = 0; r < 8; ++r) {
        const int row = rowbase + mhi + r;
        if (row < n_rows) {
          float v0 = acc0[r] + bc0;
          float v1 = acc1[r] + bc1;
          if (do_relu) { v0 = fmaxf(v0, 0.f); v1 = fmaxf(v1, 0.f); }
          Y[(size_t)row * DD + col0] = v0;
          Y[(size_t)row * DD + col1] = v1;
        }
      }
    }
  }
}

// ---------------------------------------------------------------------------
// Per-layer state init: acc=0, amax=-inf, denom=0, reduction scalars=0.
// ---------------------------------------------------------------------------
__global__ void init_layer_kernel(float* __restrict__ acc, float* __restrict__ amax,
                                  float* __restrict__ denom, double* __restrict__ red,
                                  int n_acc, int n_nh) {
  const int i = blockIdx.x * blockDim.x + threadIdx.x;
  if (i < n_acc) acc[i] = 0.f;
  if (i < n_nh) { amax[i] = -__builtin_inff(); denom[i] = 0.f; }
  if (i < 2) red[i] = 0.0;
}

__device__ __forceinline__ void atomicMaxF(float* addr, float v) {
  // monotone in both encodings: ints for >=0, uints (reversed) for <0
  if (v >= 0.f) atomicMax((int*)addr, __float_as_int(v));
  else          atomicMin((unsigned int*)addr, __float_as_uint(v));
}

// ---------------------------------------------------------------------------
// Wave per edge: alpha[e,h] = sum_c att[h,c]*lrelu(xl[src]+xr[dst]+ea)
// ea built on the fly from one-hot edge_type (We[t] + feat*We[3]).
// Then segment-max into amax[dst,h] via float atomic max.
// ---------------------------------------------------------------------------
__global__ void edge_alpha_kernel(const float* __restrict__ xl, const float* __restrict__ xr,
                                  const int* __restrict__ src, const int* __restrict__ dst,
                                  const int* __restrict__ etype, const float* __restrict__ efeat,
                                  const float* __restrict__ We, const float* __restrict__ att,
                                  float* __restrict__ alpha, float* __restrict__ amax, int E) {
  const int lane = threadIdx.x & 31;
  const int e = blockIdx.x * 8 + (threadIdx.x >> 5);
  if (e >= E) return;
  const int s = src[e], d = dst[e], t = etype[e];
  const float f = efeat[e];
  const float* __restrict__ xls = xl + (size_t)s * DD;
  const float* __restrict__ xrd = xr + (size_t)d * DD;
  float p[HH] = {0.f, 0.f, 0.f, 0.f};
#pragma unroll
  for (int j = 0; j < 8; ++j) {
    const int ch = lane + 32 * j;                     // head = ch/64 = j>>1
    const float ea = We[t * DD + ch] + f * We[3 * DD + ch];
    float v = xls[ch] + xrd[ch] + ea;
    v = (v >= 0.f) ? v : NEG_SLOPE * v;
    p[j >> 1] += att[ch] * v;
  }
#pragma unroll
  for (int off = 16; off >= 1; off >>= 1) {
#pragma unroll
    for (int h = 0; h < HH; ++h) p[h] += __shfl_xor(p[h], off, 32);
  }
  if (lane == 0) {
#pragma unroll
    for (int h = 0; h < HH; ++h) {
      alpha[e * HH + h] = p[h];
      atomicMaxF(&amax[d * HH + h], p[h]);
    }
  }
}

// ---------------------------------------------------------------------------
// ex = exp(alpha - amax[dst]); denom[dst] += ex  (alpha overwritten with ex)
// ---------------------------------------------------------------------------
__global__ void edge_softmax_kernel(float* __restrict__ alpha, const float* __restrict__ amax,
                                    const int* __restrict__ dst, float* __restrict__ denom,
                                    int E) {
  const int i = blockIdx.x * blockDim.x + threadIdx.x;
  if (i >= E * HH) return;
  const int e = i >> 2, h = i & 3;
  const int d = dst[e];
  const float ex = __expf(alpha[i] - amax[d * HH + h]);
  alpha[i] = ex;
  atomicAdd(&denom[d * HH + h], ex);
}

// ---------------------------------------------------------------------------
// Wave per edge: acc[dst] += xl[src] * (ex/denom[dst]) per head
// ---------------------------------------------------------------------------
__global__ void edge_message_kernel(const float* __restrict__ exbuf,
                                    const float* __restrict__ denom,
                                    const float* __restrict__ xl,
                                    const int* __restrict__ src, const int* __restrict__ dst,
                                    float* __restrict__ acc, int E) {
  const int lane = threadIdx.x & 31;
  const int e = blockIdx.x * 8 + (threadIdx.x >> 5);
  if (e >= E) return;
  const int s = src[e], d = dst[e];
  float a[HH];
#pragma unroll
  for (int h = 0; h < HH; ++h) a[h] = exbuf[e * HH + h] / denom[d * HH + h];
  const float* __restrict__ xls = xl + (size_t)s * DD;
  float* __restrict__ ad = acc + (size_t)d * DD;
#pragma unroll
  for (int j = 0; j < 8; ++j) {
    const int ch = lane + 32 * j;
    atomicAdd(&ad[ch], xls[ch] * a[j >> 1]);
  }
}

// ---------------------------------------------------------------------------
// r = relu(acc + bo); accumulate global sum / sumsq (double) for graph-LN.
// ---------------------------------------------------------------------------
__global__ void node_reduce_kernel(const float* __restrict__ acc, const float* __restrict__ bo,
                                   float* __restrict__ rbuf, double* __restrict__ red,
                                   int total) {
  __shared__ double ssum[8], ssq[8];
  const int i = blockIdx.x * blockDim.x + threadIdx.x;
  double ls = 0.0, lq = 0.0;
  if (i < total) {
    const int col = i & (DD - 1);
    const float r = fmaxf(acc[i] + bo[col], 0.f);
    rbuf[i] = r;
    ls = (double)r;
    lq = (double)r * (double)r;
  }
#pragma unroll
  for (int off = 16; off >= 1; off >>= 1) {
    ls += __shfl_xor(ls, off, 32);
    lq += __shfl_xor(lq, off, 32);
  }
  const int lane = threadIdx.x & 31, wave = threadIdx.x >> 5;
  if (lane == 0) { ssum[wave] = ls; ssq[wave] = lq; }
  __syncthreads();
  if (threadIdx.x == 0) {
    double ts = 0.0, tq = 0.0;
#pragma unroll
    for (int w = 0; w < 8; ++w) { ts += ssum[w]; tq += ssq[w]; }
    atomicAdd(&red[0], ts);
    atomicAdd(&red[1], tq);
  }
}

__global__ void ln_stats_kernel(const double* __restrict__ red, float* __restrict__ stats,
                                double inv_count) {
  if (blockIdx.x == 0 && threadIdx.x == 0) {
    const double mu  = red[0] * inv_count;
    const double var = red[1] * inv_count - mu * mu;
    stats[0] = (float)mu;
    stats[1] = (float)(1.0 / sqrt(var + (double)LN_EPS));
  }
}

__global__ void apply_ln_kernel(const float* __restrict__ rbuf, const float* __restrict__ gamma,
                                const float* __restrict__ beta, const float* __restrict__ stats,
                                float* __restrict__ out, int total) {
  const int i = blockIdx.x * blockDim.x + threadIdx.x;
  if (i >= total) return;
  const int col = i & (DD - 1);
  out[i] = (rbuf[i] - stats[0]) * stats[1] * gamma[col] + beta[col];
}

// ---------------------------------------------------------------------------
extern "C" void kernel_launch(void* const* d_in, const int* in_sizes, int n_in,
                              void* d_out, int out_size, void* d_ws, size_t ws_size,
                              hipStream_t stream) {
  const float* x     = (const float*)d_in[0];
  const int*   eidx  = (const int*)  d_in[1];
  const int*   etype = (const int*)  d_in[2];
  const float* efeat = (const float*)d_in[3];
  const float* W0    = (const float*)d_in[4];
  const float* b0    = (const float*)d_in[5];
  const float* Wl    = (const float*)d_in[6];
  const float* bl    = (const float*)d_in[7];
  const float* Wr    = (const float*)d_in[8];
  const float* br    = (const float*)d_in[9];
  const float* We    = (const float*)d_in[10];
  const float* att   = (const float*)d_in[11];
  const float* bo    = (const float*)d_in[12];
  const float* gamma = (const float*)d_in[13];
  const float* beta  = (const float*)d_in[14];

  const int N = in_sizes[0] / 32;   // x is [N,32]
  const int E = in_sizes[2];        // edge_type is [E]
  const int* src = eidx;            // edge_index[0]
  const int* dst = eidx + E;        // edge_index[1]
  const int ND = N * DD;

  // workspace carving (256B aligned)
  char* ws = (char*)d_ws;
  size_t off = 0;
  auto carve = [&](size_t bytes) -> void* {
    void* p = (void*)(ws + off);
    off += (bytes + 255) & ~(size_t)255;
    return p;
  };
  float*  hbuf  = (float*) carve((size_t)ND * 4);
  float*  xlb   = (float*) carve((size_t)ND * 4);
  float*  xrb   = (float*) carve((size_t)ND * 4);
  float*  accb  = (float*) carve((size_t)ND * 4);
  float*  alpha = (float*) carve((size_t)E * HH * 4);
  float*  amax  = (float*) carve((size_t)N * HH * 4);
  float*  denom = (float*) carve((size_t)N * HH * 4);
  double* red   = (double*)carve(64);
  float*  stats = (float*) carve(64);
  __bf16* WpA   = (__bf16*)carve((size_t)16 * 8 * 32 * 16 * 2);
  __bf16* WpB   = (__bf16*)carve((size_t)16 * 8 * 32 * 16 * 2);
  (void)ws_size; (void)n_in; (void)out_size;

  const int gemm_blocks = ((N + 15) / 16 + 7) / 8;   // 8 waves (16 rows each)/block
  const int nd_blocks   = (ND + 255) / 256;
  const int edge_blocks = (E + 7) / 8;
  const int eh_blocks   = (E * HH + 255) / 256;
  const double inv_cnt  = 1.0 / (double)ND;

  // pre-encoder: h = relu(x @ W0 + b0), K = 32
  pack_w_kernel<<<(16 * 1 * 32 + 255) / 256, 256, 0, stream>>>(W0, WpA, 32);
  gemm_wmma_kernel<1><<<gemm_blocks, 256, 0, stream>>>(x, 32, WpA, b0, hbuf, N, 1);

  for (int l = 0; l < 3; ++l) {
    const float* Wl_l = Wl + (size_t)l * DD * DD;
    const float* Wr_l = Wr + (size_t)l * DD * DD;

    pack_w_kernel<<<(16 * 8 * 32 + 255) / 256, 256, 0, stream>>>(Wl_l, WpA, DD);
    pack_w_kernel<<<(16 * 8 * 32 + 255) / 256, 256, 0, stream>>>(Wr_l, WpB, DD);
    gemm_wmma_kernel<8><<<gemm_blocks, 256, 0, stream>>>(hbuf, DD, WpA, bl + l * DD, xlb, N, 0);
    gemm_wmma_kernel<8><<<gemm_blocks, 256, 0, stream>>>(hbuf, DD, WpB, br + l * DD, xrb, N, 0);

    init_layer_kernel<<<nd_blocks, 256, 0, stream>>>(accb, amax, denom, red, ND, N * HH);

    edge_alpha_kernel<<<edge_blocks, 256, 0, stream>>>(
        xlb, xrb, src, dst, etype, efeat, We + (size_t)l * 4 * DD, att + (size_t)l * DD,
        alpha, amax, E);
    edge_softmax_kernel<<<eh_blocks, 256, 0, stream>>>(alpha, amax, dst, denom, E);
    edge_message_kernel<<<edge_blocks, 256, 0, stream>>>(alpha, denom, xlb, src, dst, accb, E);

    node_reduce_kernel<<<nd_blocks, 256, 0, stream>>>(accb, bo + l * DD, hbuf, red, ND);
    ln_stats_kernel<<<1, 32, 0, stream>>>(red, stats, inv_cnt);
    apply_ln_kernel<<<nd_blocks, 256, 0, stream>>>(
        hbuf, gamma, beta, stats, (l == 2) ? (float*)d_out : hbuf, ND);
  }
}
